// WeightMatrixTransformerDecoderV2_70746701300266
// MI455X (gfx1250) — matrix operand order repaired and checked by reading, not compiled, not run on hardware
//
#include <hip/hip_runtime.h>
#include <hip/hip_bf16.h>

// ---------------------------------------------------------------------------
// MI455X (gfx1250, wave32) fused kernel:
//   kv = token @ W + b ; split K/V per 32-col head group; silu+rope(K);
//   out[b, d*16+e, h] = sum_n K[b,h,n,d] * V[b,h,n,e]
// bf16 v_wmma_f32_16x16x32_bf16 for the 68.7-GFLOP GEMM.
// Block = (batch, 2 head-groups, all 256 rows): halves token L2 traffic;
// W tiles staged via global_load_async_to_lds_b128 (double buffered) so the
// 8 waves share one 4KB stage per k-chunk instead of 8 redundant loads.
// ---------------------------------------------------------------------------

typedef __attribute__((ext_vector_type(16))) __bf16 v16bf;
typedef __attribute__((ext_vector_type(8)))  float  v8f;
typedef __attribute__((ext_vector_type(4)))  int    v4i;

#define IN_DIM   512
#define OUT_COLS 16384
#define SEQ_N    256

__device__ __forceinline__ unsigned pack_bf16x2(float lo, float hi) {
    __bf16 l = (__bf16)lo, h = (__bf16)hi;
    unsigned short ls, hs;
    __builtin_memcpy(&ls, &l, 2);
    __builtin_memcpy(&hs, &h, 2);
    return (unsigned)ls | ((unsigned)hs << 16);
}

__device__ __forceinline__ v16bf load_frag16(const __bf16* p0, const __bf16* p1) {
    v16bf f;
    ((v4i*)&f)[0] = *(const v4i*)p0;   // elements 0..7  (16B)
    ((v4i*)&f)[1] = *(const v4i*)p1;   // elements 8..15 (16B)
    return f;
}

// Async 16B copy global -> LDS (CDNA5 ASYNCcnt path). ldsoff = LDS byte
// address (low 32 bits of generic pointer), goff = byte offset vs saddr base.
__device__ __forceinline__ void async_copy16(unsigned ldsoff, unsigned goff,
                                             const void* sbase) {
    asm volatile("global_load_async_to_lds_b128 %0, %1, %2"
                 :
                 : "v"(ldsoff), "v"(goff), "s"(sbase)
                 : "memory");
}

// ---------------- token fp32 -> bf16 (4 elems / thread) ---------------------
__global__ __launch_bounds__(256) void cvt_tok_kernel(const float* __restrict__ x,
                                                      __bf16* __restrict__ y) {
    int i = (blockIdx.x * 256 + threadIdx.x) * 4;
    float4 f = *(const float4*)(x + i);
    unsigned lo = pack_bf16x2(f.x, f.y);
    unsigned hi = pack_bf16x2(f.z, f.w);
    *(uint2*)(y + i) = make_uint2(lo, hi);
}

// -------- W (512 x 16384) fp32 -> Wt (16384 x 512) bf16, LDS-tiled ----------
__global__ __launch_bounds__(256) void transpose_w_kernel(const float* __restrict__ W,
                                                          __bf16* __restrict__ Wt) {
    __shared__ float tile[32][33];
    const int c0 = blockIdx.x * 32;
    const int k0 = blockIdx.y * 32;
    const int tx = threadIdx.x & 31;
    const int ty = threadIdx.x >> 5;
#pragma unroll
    for (int j = 0; j < 4; ++j) {
        int k = ty + j * 8;
        tile[k][tx] = W[(size_t)(k0 + k) * OUT_COLS + c0 + tx];
    }
    __syncthreads();
#pragma unroll
    for (int j = 0; j < 4; ++j) {
        int cc = ty + j * 8;
        Wt[(size_t)(c0 + cc) * IN_DIM + k0 + tx] = (__bf16)tile[tx][cc];
    }
}

// ---------------------------- fused main kernel -----------------------------
// grid = (256 h-pairs, 16 batches), block = 256 threads = 8 waves.
__global__ __launch_bounds__(256) void fused_kernel(const __bf16* __restrict__ tok,
                                                    const __bf16* __restrict__ wt,
                                                    const float* __restrict__ bias,
                                                    float* __restrict__ out) {
    __shared__ __align__(16) unsigned short wstage[2][64 * 32];   // 2 x 4KB W dbl-buf
    __shared__ __align__(16) unsigned short k_lds[2][8 * 512];    // [hg][wave][d][n]
    __shared__ __align__(16) unsigned short v_lds[2][8 * 512];    // [hg][wave][e][n]
    __shared__ float wm[2][256];                                  // 16x16 per hg

    const int h0    = blockIdx.x * 2;        // first head group of this block
    const int bb    = blockIdx.y;            // batch 0..15
    const int t     = threadIdx.x;
    const int lane  = t & 31;
    const int w     = t >> 5;                // wave id 0..7
    const int lhalf = lane >> 4;             // 0 | 1
    const int lmod  = lane & 15;

    wm[0][t] = 0.0f;
    wm[1][t] = 0.0f;

    // W staging addresses: thread t copies 16B -> cols 0..63, k-quarter t&3
    const int      colL  = t >> 2;                 // block-local column 0..63
    const int      koff8 = (t & 3) * 8;            // 0,8,16,24
    const unsigned gbase = (unsigned)(((h0 * 32 + colL) * IN_DIM + koff8) * 2);
    const unsigned lbase0 = (unsigned)(uintptr_t)&wstage[0][colL * 32 + koff8];
    const unsigned lbase1 = (unsigned)(uintptr_t)&wstage[1][colL * 32 + koff8];

    // A-fragment pointers (token rows 32w..32w+31)
    const __bf16* tb  = tok + (size_t)bb * SEQ_N * IN_DIM;
    const int row0    = (2 * w) * 16 + lmod;
    const __bf16* ta0 = tb + (size_t)row0 * IN_DIM + (lhalf ? 8 : 0);
    const __bf16* ta1 = ta0 + (size_t)16 * IN_DIM;

    v8f cK[2][2] = {{{}, {}}, {{}, {}}};   // [hg][mtile]
    v8f cV[2][2] = {{{}, {}}, {{}, {}}};

    // prologue: stage chunk 0
    async_copy16(lbase0, gbase, (const void*)wt);

    for (int kc = 0; kc < 16; ++kc) {
        const int k0 = kc * 32;
        if (kc + 1 < 16) {
            async_copy16((kc & 1) ? lbase0 : lbase1,
                         gbase + (unsigned)((k0 + 32) * 2), (const void*)wt);
            asm volatile("s_wait_asynccnt 0x1" ::: "memory");
        } else {
            asm volatile("s_wait_asynccnt 0x0" ::: "memory");
        }
        __syncthreads();   // buf[kc&1] fully staged by all waves

        const unsigned short* ws = &wstage[kc & 1][0];
        v16bf a0 = load_frag16(ta0 + k0, ta0 + k0 + 16);
        v16bf a1 = load_frag16(ta1 + k0, ta1 + k0 + 16);
#pragma unroll
        for (int hg = 0; hg < 2; ++hg) {
            const unsigned short* pk = ws + (hg * 32 + lmod) * 32 + (lhalf ? 8 : 0);
            const unsigned short* pv = pk + 16 * 32;
            v16bf bK = load_frag16((const __bf16*)pk, (const __bf16*)(pk + 16));
            v16bf bV = load_frag16((const __bf16*)pv, (const __bf16*)(pv + 16));
            cK[hg][0] = __builtin_amdgcn_wmma_f32_16x16x32_bf16(false, a0, false, bK, (short)0, cK[hg][0], false, false);
            cV[hg][0] = __builtin_amdgcn_wmma_f32_16x16x32_bf16(false, a0, false, bV, (short)0, cV[hg][0], false, false);
            cK[hg][1] = __builtin_amdgcn_wmma_f32_16x16x32_bf16(false, a1, false, bK, (short)0, cK[hg][1], false, false);
            cV[hg][1] = __builtin_amdgcn_wmma_f32_16x16x32_bf16(false, a1, false, bV, (short)0, cV[hg][1], false, false);
        }
        __syncthreads();   // protect buf[kc&1] against stage(kc+2) overwrite
    }

    // ---- epilogue: bias, silu, rope on K; bias on V; stage to LDS ----------
    const int d   = lmod;
    const float sgn = (d & 1) ? 1.0f : -1.0f;
#pragma unroll
    for (int hg = 0; hg < 2; ++hg) {
        const int   h     = h0 + hg;
        const float biasK = bias[h * 32 + lmod];
        const float biasV = bias[h * 32 + 16 + lmod];
        const float freq  = __expf(-0.00224861828f * (float)(h * 8 + (d >> 1)));
#pragma unroll
        for (int mt = 0; mt < 2; ++mt) {
            v8f kacc = cK[hg][mt];
            v8f vacc = cV[hg][mt];
            const int nbase = 32 * w + 16 * mt;
            float kv[8];
#pragma unroll
            for (int r = 0; r < 8; ++r) {
                float x = kacc[r] + biasK;
                x = x / (1.0f + __expf(-x));              // silu
                float nb = __shfl_xor(x, 1, 32);          // rope partner col d^1
                const int n = nbase + r + 8 * lhalf;
                float s, c;
                __sincosf(freq * (float)(SEQ_N - 1 - n), &s, &c);
                kv[r] = x * c + sgn * nb * s;
            }
            unsigned short* kb = &k_lds[hg][w * 512 + d * 32 + 16 * mt + 8 * lhalf];
            unsigned short* vb = &v_lds[hg][w * 512 + d * 32 + 16 * mt + 8 * lhalf];
#pragma unroll
            for (int r = 0; r < 8; r += 2) {
                *(unsigned*)(kb + r) = pack_bf16x2(kv[r], kv[r + 1]);
                *(unsigned*)(vb + r) = pack_bf16x2(vacc[r] + biasV, vacc[r + 1] + biasV);
            }
        }
    }
    __syncthreads();

    // ---- phase 2: per-wave K^T(16x32) @ V(32x16), ds_add_f32 reduction -----
#pragma unroll
    for (int hg = 0; hg < 2; ++hg) {
        const unsigned short* kw = &k_lds[hg][w * 512];
        const unsigned short* vw = &v_lds[hg][w * 512];
        v16bf fa = load_frag16((const __bf16*)(kw + lmod * 32 + 8 * lhalf),
                               (const __bf16*)(kw + lmod * 32 + 8 * lhalf + 16));
        v16bf fb = load_frag16((const __bf16*)(vw + lmod * 32 + 8 * lhalf),
                               (const __bf16*)(vw + lmod * 32 + 8 * lhalf + 16));
        v8f p = {};
        p = __builtin_amdgcn_wmma_f32_16x16x32_bf16(false, fa, false, fb, (short)0, p, false, false);
#pragma unroll
        for (int r = 0; r < 8; ++r) {
            atomicAdd(&wm[hg][(r + 8 * lhalf) * 16 + lmod], p[r]);
        }
    }
    __syncthreads();

    out[((size_t)bb * 256 + t) * 512 + h0 + 0] = wm[0][t];
    out[((size_t)bb * 256 + t) * 512 + h0 + 1] = wm[1][t];
}

// ---------------------------------------------------------------------------
extern "C" void kernel_launch(void* const* d_in, const int* in_sizes, int n_in,
                              void* d_out, int out_size, void* d_ws, size_t ws_size,
                              hipStream_t stream) {
    const float* token = (const float*)d_in[0];   // (16,256,512)
    const float* Wproj = (const float*)d_in[1];   // (512,16384)
    const float* bproj = (const float*)d_in[2];   // (16384,)
    float* out = (float*)d_out;                   // (16,256,512)

    __bf16* tok_bf = (__bf16*)d_ws;                               // 4,194,304 B
    __bf16* wt     = (__bf16*)((char*)d_ws + 4u * 1024u * 1024u); // 16,777,216 B

    cvt_tok_kernel<<<2048, 256, 0, stream>>>(token, tok_bf);
    transpose_w_kernel<<<dim3(512, 16), 256, 0, stream>>>(Wproj, wt);
    fused_kernel<<<dim3(256, 16), 256, 0, stream>>>(tok_bf, wt, bproj, out);
    (void)in_sizes; (void)n_in; (void)out_size; (void)ws_size;
}
